// NonLocalBlock2_33320356283059
// MI455X (gfx1250) — matrix operand order repaired and checked by reading, not compiled, not run on hardware
//
#include <hip/hip_runtime.h>

#define B_ 4
#define C_ 256
#define N_ 4096
#define CK_ 32
#define CV_ 128

typedef __attribute__((ext_vector_type(16))) __bf16 v16bf;
typedef __attribute__((ext_vector_type(8)))  float  v8f;
typedef unsigned int   u32;
typedef unsigned short u16;

__device__ __forceinline__ __bf16 f2bf(float f) {
    u32 u = __builtin_bit_cast(u32, f);
    u32 r = u + 0x7FFFu + ((u >> 16) & 1u);      // round-to-nearest-even
    u16 s = (u16)(r >> 16);
    return __builtin_bit_cast(__bf16, s);
}
__device__ __forceinline__ float bf2f(__bf16 h) {
    u16 s = __builtin_bit_cast(u16, h);
    u32 u = ((u32)s) << 16;
    return __builtin_bit_cast(float, u);
}

// ---- WMMA fragment builders (CDNA5 16-bit layouts, CDNA5 ISA) ----
// A (16x32, MxK): lane m = lane&15; lanes0-15: V0-3 K=0..7, V4-7 K=16..23;
// lanes16-31: +8.  Source: row-major [m][k], pairs contiguous -> u32 loads.
template <typename PT>
__device__ __forceinline__ v16bf load_frag_a(PT base, int ld) {
    int lane = threadIdx.x & 31;
    int m  = lane & 15;
    int kq = (lane & 16) ? 8 : 0;
    PT row = base + m * ld;
    union { v16bf v; u32 u[8]; } A;
#pragma unroll
    for (int v = 0; v < 4; ++v) A.u[v] = *(const u32*)(row + 2 * v + kq);
#pragma unroll
    for (int v = 4; v < 8; ++v) A.u[v] = *(const u32*)(row + 8 + 2 * v + kq);
    return A.v;
}
// B (32x16, KxN): lane n = lane&15; lanes0-15 hold K=0..7,16..23, lanes16-31 +8,
// 2 K-values per VGPR.  Source stored TRANSPOSED: src[n][k], pairs contiguous.
template <typename PT>
__device__ __forceinline__ v16bf load_frag_b_t(PT base, int ld) {
    int lane = threadIdx.x & 31;
    int n  = lane & 15;
    int kb = (lane & 16) ? 8 : 0;
    PT row = base + n * ld + kb;
    union { v16bf v; u32 u[8]; } Bf;
#pragma unroll
    for (int v = 0; v < 8; ++v) Bf.u[v] = *(const u32*)(row + 2 * v + (v >= 4 ? 8 : 0));
    return Bf.v;
}
__device__ __forceinline__ v8f wmma_bf16(v16bf a, v16bf b, v8f c) {
    return __builtin_amdgcn_wmma_f32_16x16x32_bf16(false, a, false, b, (short)0, c,
                                                   false, false);
}

// ---- kernel 0a: f32 -> bf16 weight conversion ----
__global__ void k_cvt(const float* __restrict__ src, __bf16* __restrict__ dst, int n) {
    int i = blockIdx.x * blockDim.x + threadIdx.x;
    if (i < n) dst[i] = f2bf(src[i]);
}

// ---- kernel 0b: x [B][C][N] f32 -> xT [B][N][C] bf16 ----
__global__ void k_xT(const float* __restrict__ x, __bf16* __restrict__ xT) {
    int i = blockIdx.x * blockDim.x + threadIdx.x;
    if (i >= B_ * C_ * N_) return;
    int n = i % N_;
    int c = (i / N_) % C_;
    int b = i / (N_ * C_);
    xT[((size_t)b * N_ + n) * C_ + c] = f2bf(x[i]);
}

// ---- kernel 1: projection GEMM  out[M,N] = W[M,256] @ xT^T + bias ----
// transposed=1: store out as [B][N][M] (for theta/phi); else [B][M][N] (g).
__global__ __launch_bounds__(128) void k_proj(const __bf16* __restrict__ W,
                                              const float* __restrict__ bias,
                                              const __bf16* __restrict__ xT,
                                              __bf16* __restrict__ out, int M,
                                              int transposed) {
    int b    = blockIdx.z;
    int wv   = threadIdx.x >> 5;
    int lane = threadIdx.x & 31;
    int n0   = blockIdx.x * 64 + wv * 16;
    int m0   = blockIdx.y * 16;
    const __bf16* xTb = xT + (size_t)b * N_ * C_;
    v8f acc = {};
    for (int kc = 0; kc < C_; kc += 32) {
        v16bf a  = load_frag_a(W + (size_t)m0 * C_ + kc, C_);
        v16bf bb = load_frag_b_t(xTb + (size_t)n0 * C_ + kc, C_);
        acc = wmma_bf16(a, bb, acc);
    }
    int n  = lane & 15;
    int rb = (lane & 16) ? 8 : 0;
#pragma unroll
    for (int r = 0; r < 8; ++r) {
        int   m = m0 + rb + r;
        float v = acc[r] + bias[m];
        if (transposed)
            out[((size_t)b * N_ + (n0 + n)) * M + m] = f2bf(v);
        else
            out[((size_t)b * M + m) * N_ + (n0 + n)] = f2bf(v);
    }
}

// ---- kernel 2: per-column softmax stats over score matrix (recomputed) ----
__global__ __launch_bounds__(128) void k_colstats(const __bf16* __restrict__ thT,
                                                  const __bf16* __restrict__ phT,
                                                  float* __restrict__ cmax,
                                                  float* __restrict__ csum) {
    int b    = blockIdx.y;
    int wv   = threadIdx.x >> 5;
    int lane = threadIdx.x & 31;
    int j0   = blockIdx.x * 64 + wv * 16;
    const __bf16* th = thT + (size_t)b * N_ * CK_;
    const __bf16* ph = phT + (size_t)b * N_ * CK_;
    v16bf bfrag = load_frag_b_t(ph + (size_t)j0 * CK_, CK_);   // loop-invariant
    float m = -3.0e38f, l = 0.f;
    for (int i0 = 0; i0 < N_; i0 += 16) {
        v16bf a = load_frag_a(th + (size_t)i0 * CK_, CK_);
        v8f z = {};
        v8f s = wmma_bf16(a, bfrag, z);
        float tmax = s[0];
#pragma unroll
        for (int r = 1; r < 8; ++r) tmax = fmaxf(tmax, s[r]);
        float nm    = fmaxf(m, tmax);
        float scale = __expf(m - nm);
        float add   = 0.f;
#pragma unroll
        for (int r = 0; r < 8; ++r) add += __expf(s[r] - nm);
        l = l * scale + add;
        m = nm;
    }
    // combine row-halves (lanes l and l+16 hold same column j)
    float mo = __shfl_xor(m, 16, 32);
    float lo = __shfl_xor(l, 16, 32);
    float M2 = fmaxf(m, mo);
    float L2 = l * __expf(m - M2) + lo * __expf(mo - M2);
    if (lane < 16) {
        cmax[(size_t)b * N_ + j0 + lane] = M2;
        csum[(size_t)b * N_ + j0 + lane] = L2;
    }
}

// ---- kernel 2b: fold 1/colsum into g ----
__global__ void k_gl(const __bf16* __restrict__ g, const float* __restrict__ csum,
                     __bf16* __restrict__ gl) {
    int i = blockIdx.x * blockDim.x + threadIdx.x;
    if (i >= B_ * CV_ * N_) return;
    int j = i % N_;
    int b = i / (N_ * CV_);
    gl[i] = f2bf(bf2f(g[i]) / csum[(size_t)b * N_ + j]);
}

// ---- kernel 3: fused scores+exp+PV:  o_preT[i][c] = sum_j P(i,j)*gl[c,j] ----
__global__ __launch_bounds__(128) void k_attn(const __bf16* __restrict__ thT,
                                              const __bf16* __restrict__ phT,
                                              const __bf16* __restrict__ gl,
                                              const float* __restrict__ cmax,
                                              __bf16* __restrict__ opT) {
    __shared__ __bf16 Ps[4][16][34];   // PsT[i][j], padded rows vs bank conflicts
    int b    = blockIdx.y;
    int i0   = blockIdx.x * 16;
    int wv   = threadIdx.x >> 5;
    int lane = threadIdx.x & 31;
    int n    = lane & 15;
    int rb   = (lane & 16) ? 8 : 0;
    const __bf16* th  = thT + (size_t)b * N_ * CK_;
    const __bf16* ph  = phT + (size_t)b * N_ * CK_;
    const __bf16* glb = gl + (size_t)b * CV_ * N_;
    const float*  cm  = cmax + (size_t)b * N_;
    v16bf afrag = load_frag_a(th + (size_t)i0 * CK_, CK_);  // theta tile, invariant
    v8f acc0 = {}, acc1 = {};
    int c0 = wv * 32;
    for (int jb = 0; jb < N_; jb += 128) {
        // phase 1: this wave computes P for its 32 columns
#pragma unroll
        for (int jt = 0; jt < 2; ++jt) {
            int   jg0 = jb + wv * 32 + jt * 16;
            v16bf bph = load_frag_b_t(ph + (size_t)jg0 * CK_, CK_);
            v8f z = {};
            v8f s = wmma_bf16(afrag, bph, z);
            float mj   = cm[jg0 + n];
            int   jloc = jt * 16 + n;
#pragma unroll
            for (int r = 0; r < 8; ++r)
                Ps[wv][rb + r][jloc] = f2bf(__expf(s[r] - mj));
        }
        __syncthreads();
        // phase 2: every wave consumes all 4 P chunks against its 2 c-tiles
#pragma unroll
        for (int w2 = 0; w2 < 4; ++w2) {
            int   jk = jb + w2 * 32;
            v16bf bp = load_frag_b_t(&Ps[w2][0][0], 34);
            v16bf a0 = load_frag_a(glb + (size_t)c0 * N_ + jk, N_);
            acc0 = wmma_bf16(a0, bp, acc0);
            v16bf a1 = load_frag_a(glb + (size_t)(c0 + 16) * N_ + jk, N_);
            acc1 = wmma_bf16(a1, bp, acc1);
        }
        __syncthreads();
    }
#pragma unroll
    for (int r = 0; r < 8; ++r) {
        size_t o = ((size_t)b * N_ + i0 + n) * CV_;
        opT[o + c0 + rb + r]      = f2bf(acc0[r]);
        opT[o + c0 + 16 + rb + r] = f2bf(acc1[r]);
    }
}

// ---- kernel 4: o = o_w @ o_pre, out = gamma*o + x ----
__global__ __launch_bounds__(128) void k_out(const __bf16* __restrict__ wO,
                                             const float* __restrict__ ob,
                                             const __bf16* __restrict__ opT,
                                             const float* __restrict__ x,
                                             const float* __restrict__ gamma,
                                             float* __restrict__ out) {
    int b    = blockIdx.z;
    int wv   = threadIdx.x >> 5;
    int lane = threadIdx.x & 31;
    int n0   = blockIdx.x * 64 + wv * 16;
    int m0   = blockIdx.y * 16;
    const __bf16* opb = opT + (size_t)b * N_ * CV_;
    v8f acc = {};
#pragma unroll
    for (int kc = 0; kc < CV_; kc += 32) {
        v16bf a  = load_frag_a(wO + (size_t)m0 * CV_ + kc, CV_);
        v16bf bb = load_frag_b_t(opb + (size_t)n0 * CV_ + kc, CV_);
        acc = wmma_bf16(a, bb, acc);
    }
    float gm = gamma[0];
    int n  = lane & 15;
    int rb = (lane & 16) ? 8 : 0;
#pragma unroll
    for (int r = 0; r < 8; ++r) {
        int    m  = m0 + rb + r;
        size_t xi = ((size_t)b * C_ + m) * N_ + n0 + n;
        out[xi] = gm * (acc[r] + ob[m]) + x[xi];
    }
}

extern "C" void kernel_launch(void* const* d_in, const int* in_sizes, int n_in,
                              void* d_out, int out_size, void* d_ws, size_t ws_size,
                              hipStream_t stream) {
    const float* x       = (const float*)d_in[0];
    const float* theta_w = (const float*)d_in[1];
    const float* theta_b = (const float*)d_in[2];
    const float* phi_w   = (const float*)d_in[3];
    const float* phi_b   = (const float*)d_in[4];
    const float* g_w     = (const float*)d_in[5];
    const float* g_b     = (const float*)d_in[6];
    const float* o_w     = (const float*)d_in[7];
    const float* o_b     = (const float*)d_in[8];
    const float* gamma   = (const float*)d_in[9];
    float* out = (float*)d_out;

    // workspace carve-up (256B aligned)
    char*  base = (char*)d_ws;
    size_t off  = 0;
    auto alloc = [&](size_t bytes) -> char* {
        char* p = base + off;
        off += (bytes + 255) & ~(size_t)255;
        return p;
    };
    __bf16* wT   = (__bf16*)alloc((size_t)CK_ * C_ * 2);
    __bf16* wP   = (__bf16*)alloc((size_t)CK_ * C_ * 2);
    __bf16* wG   = (__bf16*)alloc((size_t)CV_ * C_ * 2);
    __bf16* wO   = (__bf16*)alloc((size_t)C_ * CV_ * 2);
    __bf16* xT   = (__bf16*)alloc((size_t)B_ * N_ * C_ * 2);
    __bf16* thT  = (__bf16*)alloc((size_t)B_ * N_ * CK_ * 2);
    __bf16* phT  = (__bf16*)alloc((size_t)B_ * N_ * CK_ * 2);
    __bf16* gbuf = (__bf16*)alloc((size_t)B_ * CV_ * N_ * 2);
    __bf16* glb  = (__bf16*)alloc((size_t)B_ * CV_ * N_ * 2);
    float*  cmax = (float*)alloc((size_t)B_ * N_ * 4);
    float*  csum = (float*)alloc((size_t)B_ * N_ * 4);
    __bf16* opT  = (__bf16*)alloc((size_t)B_ * N_ * CV_ * 2);

    // 0a: weights -> bf16
    k_cvt<<<(CK_ * C_ + 255) / 256, 256, 0, stream>>>(theta_w, wT, CK_ * C_);
    k_cvt<<<(CK_ * C_ + 255) / 256, 256, 0, stream>>>(phi_w, wP, CK_ * C_);
    k_cvt<<<(CV_ * C_ + 255) / 256, 256, 0, stream>>>(g_w, wG, CV_ * C_);
    k_cvt<<<(C_ * CV_ + 255) / 256, 256, 0, stream>>>(o_w, wO, C_ * CV_);
    // 0b: x -> xT bf16
    k_xT<<<(B_ * C_ * N_ + 255) / 256, 256, 0, stream>>>(x, xT);
    // 1: projections (theta/phi stored [N][32], g stored [128][N])
    k_proj<<<dim3(N_ / 64, CK_ / 16, B_), 128, 0, stream>>>(wT, theta_b, xT, thT, CK_, 1);
    k_proj<<<dim3(N_ / 64, CK_ / 16, B_), 128, 0, stream>>>(wP, phi_b, xT, phT, CK_, 1);
    k_proj<<<dim3(N_ / 64, CV_ / 16, B_), 128, 0, stream>>>(wG, g_b, xT, gbuf, CV_, 0);
    // 2: column softmax statistics
    k_colstats<<<dim3(N_ / 64, B_), 128, 0, stream>>>(thT, phT, cmax, csum);
    // 2b: gl = g / colsum
    k_gl<<<(B_ * CV_ * N_ + 255) / 256, 256, 0, stream>>>(gbuf, csum, glb);
    // 3: fused attention (scores -> exp -> PV)
    k_attn<<<dim3(N_ / 16, B_), 128, 0, stream>>>(thT, phT, glb, cmax, opT);
    // 4: output projection + residual
    k_out<<<dim3(N_ / 64, C_ / 16, B_), 128, 0, stream>>>(wO, o_b, opT, x, gamma, out);
}
